// GroupedQueryAttention_29205777613577
// MI455X (gfx1250) — matrix operand order repaired and checked by reading, not compiled, not run on hardware
//
#include <hip/hip_runtime.h>
#include <hip/hip_bf16.h>
#include <cstdint>

#define EMB  2048
#define NH_Q 32
#define NKV  8
#define HD   64
#define SEQ  2048
#define GRP  (NH_Q / NKV)

typedef __attribute__((ext_vector_type(16))) __bf16 v16bf;
typedef __attribute__((ext_vector_type(8)))  __bf16 v8bf;
typedef __attribute__((ext_vector_type(8)))  float  v8f;

union BF16x16 { v16bf v; v8bf h[2]; };

static __device__ __forceinline__ v8f zero8() {
  v8f z = {0.f, 0.f, 0.f, 0.f, 0.f, 0.f, 0.f, 0.f};
  return z;
}

// ---------------------------------------------------------------------------
// f32 -> bf16 plain convert
// ---------------------------------------------------------------------------
__global__ void f32_to_bf16(const float* __restrict__ in,
                            __bf16* __restrict__ out, int n) {
  int i = blockIdx.x * blockDim.x + threadIdx.x;
  if (i < n) out[i] = (__bf16)in[i];
}

// f32 [Kd][Nd] row-major  ->  bf16 [Nd][Kd] row-major (transpose)
__global__ void f32_to_bf16_T(const float* __restrict__ in,
                              __bf16* __restrict__ out, int Kd, int Nd) {
  int i = blockIdx.x * blockDim.x + threadIdx.x;
  if (i < Kd * Nd) {
    int k = i / Nd, n = i - k * Nd;
    out[(size_t)n * Kd + k] = (__bf16)in[i];
  }
}

// V f32 [S][KV*HD] -> bf16 Vt [KV][HD][S]  (d-major for PV WMMA B-fragments)
__global__ void v_pack_T(const float* __restrict__ in,
                         __bf16* __restrict__ out) {
  int i = blockIdx.x * blockDim.x + threadIdx.x;  // SEQ*NKV*HD
  int d  = i & (HD - 1);
  int t  = i >> 6;
  int kv = t % NKV;
  int s  = t / NKV;
  out[((size_t)kv * HD + d) * SEQ + s] = (__bf16)in[i];
}

// ---------------------------------------------------------------------------
// RMSNorm + RoPE fused, one wave32 per (s, head) row of 64 elements.
//   in  : f32 [S][NH*HD]   out : bf16 [NH][S][HD]
// ---------------------------------------------------------------------------
__global__ __launch_bounds__(256) void qk_norm_rope(
    const float* __restrict__ in, const float* __restrict__ scale,
    const float* __restrict__ cs, const float* __restrict__ sn,
    __bf16* __restrict__ out, int NH) {
  const int lane = threadIdx.x & 31;
  const int wave = threadIdx.x >> 5;
  const int row  = blockIdx.x * 8 + wave;   // row in [0, S*NH)
  const int s    = row / NH;
  const int hh   = row - s * NH;

  const float* xr = in + ((size_t)s * NH + hh) * HD;
  float x1 = xr[lane];
  float x2 = xr[lane + 32];
  float ss = x1 * x1 + x2 * x2;
#pragma unroll
  for (int off = 16; off > 0; off >>= 1) ss += __shfl_xor(ss, off, 32);
  const float rn = rsqrtf(ss * (1.0f / HD) + 1e-6f);
  const float n1 = x1 * rn * scale[lane];
  const float n2 = x2 * rn * scale[lane + 32];
  const float c1 = cs[(size_t)s * HD + lane];
  const float c2 = cs[(size_t)s * HD + lane + 32];
  const float s1 = sn[(size_t)s * HD + lane];
  const float s2 = sn[(size_t)s * HD + lane + 32];
  __bf16* orow = out + ((size_t)hh * SEQ + s) * HD;
  orow[lane]      = (__bf16)(n1 * c1 - n2 * s1);
  orow[lane + 32] = (__bf16)(n2 * c2 + n1 * s2);
}

// ---------------------------------------------------------------------------
// bf16 GEMM, v_wmma_f32_16x16x32_bf16, double-buffered CDNA5 async
// global->LDS copies (ASYNCcnt). Steady-state loop is branch-free: copies
// for tile i+1 are issued unconditionally, s_wait_asynccnt 4 drains only the
// older quartet (async loads complete in order, ISA 08 §4.1), and the last
// tile is peeled into an epilogue that drains to zero.
//   A  : bf16 [M][K] row-major
//   Bt : bf16 [N][K] row-major (B pre-transposed so fragments are contiguous)
//   C  : f32  [M][N]
// 256 threads = 8 waves; block tile 128x128; wave tile 32x64 (2x4 WMMA).
// M,N multiples of 128; K multiple of 32 with K >= 64.
// ---------------------------------------------------------------------------
__global__ __launch_bounds__(256) void gemm_bf16_wmma(
    const __bf16* __restrict__ A, const __bf16* __restrict__ Bt,
    float* __restrict__ C, int M, int N, int K) {
  __shared__ __bf16 As[2][128][32];
  __shared__ __bf16 Bs[2][128][32];

  const int tid  = threadIdx.x;
  const int lane = tid & 31;
  const int wave = tid >> 5;
  const int h    = lane >> 4;
  const int l16  = lane & 15;
  const int wrow = wave & 3;   // 4 wave rows * 32
  const int wcol = wave >> 2;  // 2 wave cols * 64
  const long bm = (long)blockIdx.y * 128;
  const long bn = (long)blockIdx.x * 128;

  // each thread owns two 16B chunks of As and two of Bs per K-step
  const int c0 = tid * 2;
  const int r0 = c0 >> 2, o0 = (c0 & 3) * 8;
  const int c1 = c0 + 1;
  const int r1 = c1 >> 2, o1 = (c1 & 3) * 8;

  // LDS byte addresses per buffer (generic pointers into the LDS aperture
  // keep the DS offset in their low 32 bits)
  const unsigned lA0[2] = {(unsigned)(size_t)&As[0][r0][o0],
                           (unsigned)(size_t)&As[1][r0][o0]};
  const unsigned lA1[2] = {(unsigned)(size_t)&As[0][r1][o1],
                           (unsigned)(size_t)&As[1][r1][o1]};
  const unsigned lB0[2] = {(unsigned)(size_t)&Bs[0][r0][o0],
                           (unsigned)(size_t)&Bs[1][r0][o0]};
  const unsigned lB1[2] = {(unsigned)(size_t)&Bs[0][r1][o1],
                           (unsigned)(size_t)&Bs[1][r1][o1]};

  auto issue_copies = [&](int buf, int kk) {
    const uint64_t ga0 = (uint64_t)&A[(bm + r0) * K + kk + o0];
    const uint64_t ga1 = (uint64_t)&A[(bm + r1) * K + kk + o1];
    const uint64_t gb0 = (uint64_t)&Bt[(bn + r0) * K + kk + o0];
    const uint64_t gb1 = (uint64_t)&Bt[(bn + r1) * K + kk + o1];
    asm volatile(
        "global_load_async_to_lds_b128 %0, %4, off\n\t"
        "global_load_async_to_lds_b128 %1, %5, off\n\t"
        "global_load_async_to_lds_b128 %2, %6, off\n\t"
        "global_load_async_to_lds_b128 %3, %7, off"
        :
        : "v"(lA0[buf]), "v"(lA1[buf]), "v"(lB0[buf]), "v"(lB1[buf]),
          "v"(ga0), "v"(ga1), "v"(gb0), "v"(gb1)
        : "memory");
  };

  v8f acc[2][4];
#pragma unroll
  for (int i = 0; i < 2; ++i)
#pragma unroll
    for (int j = 0; j < 4; ++j) acc[i][j] = zero8();

  auto compute_tile = [&](int buf) {
    v16bf af[2], bfr[4];
#pragma unroll
    for (int i = 0; i < 2; ++i) {  // A frag: lane m=l16, chunks K=h*8, 16+h*8
      BF16x16 u;
      const int m = wrow * 32 + i * 16 + l16;
      u.h[0] = *(const v8bf*)&As[buf][m][h * 8];
      u.h[1] = *(const v8bf*)&As[buf][m][16 + h * 8];
      af[i] = u.v;
    }
#pragma unroll
    for (int j = 0; j < 4; ++j) {  // B frag: lane n=l16, chunks K=h*8, 16+h*8
      BF16x16 u;
      const int n = wcol * 64 + j * 16 + l16;
      u.h[0] = *(const v8bf*)&Bs[buf][n][h * 8];
      u.h[1] = *(const v8bf*)&Bs[buf][n][16 + h * 8];
      bfr[j] = u.v;
    }
#pragma unroll
    for (int i = 0; i < 2; ++i)
#pragma unroll
      for (int j = 0; j < 4; ++j)
        acc[i][j] = __builtin_amdgcn_wmma_f32_16x16x32_bf16(
            false, af[i], false, bfr[j], (short)0, acc[i][j], false, false);
  };

  const int niter = K / 32;
  issue_copies(0, 0);  // prologue: fill buffer 0

  int it = 0;
  for (; it < niter - 1; ++it) {  // branch-free steady state
    issue_copies((it & 1) ^ 1, (it + 1) * 32);  // overlap with compute
    // speculative L2 prefetch one slab ahead (dropped silently if OOB)
    __builtin_prefetch(&A[(bm + r0) * K + (it + 2) * 32 + o0], 0, 3);
    __builtin_prefetch(&Bt[(bn + r0) * K + (it + 2) * 32 + o0], 0, 3);
    asm volatile("s_wait_asynccnt 0x4" ::: "memory");  // drain older 4 only
    __syncthreads();
    compute_tile(it & 1);
    __syncthreads();
  }
  // peeled last tile
  asm volatile("s_wait_asynccnt 0x0" ::: "memory");
  __syncthreads();
  compute_tile(it & 1);

#pragma unroll
  for (int i = 0; i < 2; ++i) {
    const long mb = bm + wrow * 32 + i * 16 + h * 8;  // D: M = r + 8h
#pragma unroll
    for (int j = 0; j < 4; ++j) {
      const long nb = bn + wcol * 64 + j * 16 + l16;  // D: N = l16
#pragma unroll
      for (int r = 0; r < 8; ++r) C[(mb + r) * N + nb] = acc[i][j][r];
    }
  }
}

// ---------------------------------------------------------------------------
// Flash attention, causal GQA. One wave per 16-query tile per head.
//   Q   : bf16 [H][S][64]     K : bf16 [KV][S][64]
//   Vt  : bf16 [KV][64][S]    ctx: bf16 [S][H*64]
// ---------------------------------------------------------------------------
__global__ __launch_bounds__(256) void gqa_flash_attn(
    const __bf16* __restrict__ Q, const __bf16* __restrict__ Kc,
    const __bf16* __restrict__ Vt, __bf16* __restrict__ ctx) {
  __shared__ __bf16 plds[8][16][32];

  const int tid  = threadIdx.x;
  const int lane = tid & 31, wave = tid >> 5;
  const int h = lane >> 4, l16 = lane & 15;
  const int head = blockIdx.y;
  const int kv   = head / GRP;
  const int q0   = (blockIdx.x * 8 + wave) * 16;

  // Q fragments (16x64, two K=32 steps) held in registers for the whole loop
  const __bf16* qrow = Q + ((size_t)head * SEQ + q0 + l16) * HD;
  v16bf qf[2];
#pragma unroll
  for (int t = 0; t < 2; ++t) {
    BF16x16 u;
    u.h[0] = *(const v8bf*)(qrow + t * 32 + h * 8);
    u.h[1] = *(const v8bf*)(qrow + t * 32 + 16 + h * 8);
    qf[t] = u.v;
  }

  float mrow[8], lrow[8];
  v8f o[4];
#pragma unroll
  for (int r = 0; r < 8; ++r) { mrow[r] = -1e30f; lrow[r] = 0.f; }
#pragma unroll
  for (int j = 0; j < 4; ++j) o[j] = zero8();

  const __bf16* kbase = Kc + (size_t)kv * SEQ * HD;
  const __bf16* vbase = Vt + (size_t)kv * HD * SEQ;

  for (int k0 = 0; k0 < q0 + 16; k0 += 32) {  // causal: keys <= q0+15
    // scores: two 16x16 N-tiles, each accumulated over d in two K=32 steps
    v8f sc[2];
#pragma unroll
    for (int j = 0; j < 2; ++j) {
      v8f a = zero8();
      const __bf16* krow = kbase + (size_t)(k0 + j * 16 + l16) * HD;
#pragma unroll
      for (int t = 0; t < 2; ++t) {
        BF16x16 u;  // B frag of K^T: lane n=key, chunks d = t*32 + h*8, t*32 + 16 + h*8
        u.h[0] = *(const v8bf*)(krow + t * 32 + h * 8);
        u.h[1] = *(const v8bf*)(krow + t * 32 + 16 + h * 8);
        a = __builtin_amdgcn_wmma_f32_16x16x32_bf16(
            false, qf[t], false, u.v, (short)0, a, false, false);
      }
      sc[j] = a;
    }

    // online softmax update, per row M = r + 8h (16-lane-half reductions)
#pragma unroll
    for (int r = 0; r < 8; ++r) {
      const int qg = q0 + r + 8 * h;
      float v0 = sc[0][r] * 0.125f;  // 1/sqrt(64)
      float v1 = sc[1][r] * 0.125f;
      if (k0 + l16 > qg)      v0 = -3.0e38f;
      if (k0 + 16 + l16 > qg) v1 = -3.0e38f;
      float mx = fmaxf(v0, v1);
#pragma unroll
      for (int off = 1; off < 16; off <<= 1)
        mx = fmaxf(mx, __shfl_xor(mx, off, 32));
      const float mnew = fmaxf(mrow[r], mx);
      const float corr = __expf(mrow[r] - mnew);
      const float p0 = __expf(v0 - mnew);
      const float p1 = __expf(v1 - mnew);
      float rs = p0 + p1;
#pragma unroll
      for (int off = 1; off < 16; off <<= 1) rs += __shfl_xor(rs, off, 32);
      lrow[r] = lrow[r] * corr + rs;
      mrow[r] = mnew;
#pragma unroll
      for (int j = 0; j < 4; ++j) o[j][r] *= corr;
      plds[wave][r + 8 * h][l16]      = (__bf16)p0;  // D -> LDS (16x32 tile)
      plds[wave][r + 8 * h][16 + l16] = (__bf16)p1;
    }

    // reload P in A-fragment layout (wave-private LDS, DS ops in-order)
    BF16x16 pu;
    pu.h[0] = *(const v8bf*)&plds[wave][l16][h * 8];
    pu.h[1] = *(const v8bf*)&plds[wave][l16][16 + h * 8];

    // O += P @ V : four 16-wide d tiles, K=32 keys per WMMA
#pragma unroll
    for (int j = 0; j < 4; ++j) {
      BF16x16 u;  // B frag of V: lane n = d, chunks key = k0 + h*8, k0 + 16 + h*8
      const __bf16* vrow = vbase + (size_t)(j * 16 + l16) * SEQ + k0 + h * 8;
      u.h[0] = *(const v8bf*)(vrow);
      u.h[1] = *(const v8bf*)(vrow + 16);
      o[j] = __builtin_amdgcn_wmma_f32_16x16x32_bf16(
          false, pu.v, false, u.v, (short)0, o[j], false, false);
    }
  }

#pragma unroll
  for (int j = 0; j < 4; ++j) {
#pragma unroll
    for (int r = 0; r < 8; ++r) {
      const int qg = q0 + r + 8 * h;
      ctx[(size_t)qg * (NH_Q * HD) + head * HD + j * 16 + l16] =
          (__bf16)(o[j][r] / lrow[r]);
    }
  }
}

// ---------------------------------------------------------------------------
extern "C" void kernel_launch(void* const* d_in, const int* in_sizes, int n_in,
                              void* d_out, int out_size, void* d_ws,
                              size_t ws_size, hipStream_t stream) {
  const float* x    = (const float*)d_in[0];
  // d_in[1] = mask (unused; causal mask computed analytically)
  const float* cosb = (const float*)d_in[2];
  const float* sinb = (const float*)d_in[3];
  const float* Wq   = (const float*)d_in[4];
  const float* Wk   = (const float*)d_in[5];
  const float* Wv   = (const float*)d_in[6];
  const float* Wo   = (const float*)d_in[7];
  const float* qs   = (const float*)d_in[8];
  const float* ks   = (const float*)d_in[9];
  float* out = (float*)d_out;

  char* p = (char*)d_ws;
  auto take = [&](size_t bytes) {
    char* r = p;
    p += (bytes + 255) & ~(size_t)255;
    return (void*)r;
  };
  __bf16* xb  = (__bf16*)take((size_t)SEQ * EMB * 2);
  __bf16* wqT = (__bf16*)take((size_t)(NH_Q * HD) * EMB * 2);
  __bf16* wkT = (__bf16*)take((size_t)(NKV * HD) * EMB * 2);
  __bf16* wvT = (__bf16*)take((size_t)(NKV * HD) * EMB * 2);
  __bf16* woT = (__bf16*)take((size_t)EMB * (NH_Q * HD) * 2);
  float*  qf  = (float*)take((size_t)SEQ * NH_Q * HD * 4);
  float*  kf  = (float*)take((size_t)SEQ * NKV * HD * 4);
  float*  vf  = (float*)take((size_t)SEQ * NKV * HD * 4);
  __bf16* qb  = (__bf16*)take((size_t)NH_Q * SEQ * HD * 2);
  __bf16* kb  = (__bf16*)take((size_t)NKV * SEQ * HD * 2);
  __bf16* vt  = (__bf16*)take((size_t)NKV * HD * SEQ * 2);
  __bf16* cb  = (__bf16*)take((size_t)SEQ * NH_Q * HD * 2);

  // 1) precision / layout conversion
  {
    int n = SEQ * EMB;
    f32_to_bf16<<<(n + 255) / 256, 256, 0, stream>>>(x, xb, n);
  }
  {
    int n = EMB * NH_Q * HD;
    f32_to_bf16_T<<<(n + 255) / 256, 256, 0, stream>>>(Wq, wqT, EMB, NH_Q * HD);
  }
  {
    int n = EMB * NKV * HD;
    f32_to_bf16_T<<<(n + 255) / 256, 256, 0, stream>>>(Wk, wkT, EMB, NKV * HD);
    f32_to_bf16_T<<<(n + 255) / 256, 256, 0, stream>>>(Wv, wvT, EMB, NKV * HD);
  }
  {
    int n = NH_Q * HD * EMB;
    f32_to_bf16_T<<<(n + 255) / 256, 256, 0, stream>>>(Wo, woT, NH_Q * HD, EMB);
  }

  // 2) QKV projections (bf16 WMMA, f32 accumulate)
  gemm_bf16_wmma<<<dim3((NH_Q * HD) / 128, SEQ / 128), 256, 0, stream>>>(
      xb, wqT, qf, SEQ, NH_Q * HD, EMB);
  gemm_bf16_wmma<<<dim3((NKV * HD) / 128, SEQ / 128), 256, 0, stream>>>(
      xb, wkT, kf, SEQ, NKV * HD, EMB);
  gemm_bf16_wmma<<<dim3((NKV * HD) / 128, SEQ / 128), 256, 0, stream>>>(
      xb, wvT, vf, SEQ, NKV * HD, EMB);

  // 3) RMSNorm + RoPE (Q, K) and V layout pack
  qk_norm_rope<<<SEQ * NH_Q / 8, 256, 0, stream>>>(qf, qs, cosb, sinb, qb, NH_Q);
  qk_norm_rope<<<SEQ * NKV / 8, 256, 0, stream>>>(kf, ks, cosb, sinb, kb, NKV);
  {
    int n = SEQ * NKV * HD;
    v_pack_T<<<(n + 255) / 256, 256, 0, stream>>>(vf, vt);
  }

  // 4) causal flash attention
  gqa_flash_attn<<<dim3(SEQ / 128, NH_Q), 256, 0, stream>>>(qb, kb, vt, cb);

  // 5) output projection -> f32
  gemm_bf16_wmma<<<dim3(EMB / 128, SEQ / 128), 256, 0, stream>>>(
      cb, woT, out, SEQ, EMB, NH_Q * HD);
}
